// LocalTransformerBlock_32830730010653
// MI455X (gfx1250) — matrix operand order repaired and checked by reading, not compiled, not run on hardware
//
#include <hip/hip_runtime.h>
#include <hip/hip_bf16.h>

typedef __bf16 bf16_t;
typedef bf16_t v16bf __attribute__((ext_vector_type(16)));
typedef float v8f __attribute__((ext_vector_type(8)));
typedef unsigned short v8u __attribute__((ext_vector_type(8)));
typedef unsigned short v16u __attribute__((ext_vector_type(16)));

#define MROWS 28672
#define HDIM  1024
#define BM 128
#define BN 128
#define BK 32
#define LDT 40  // padded LDS row stride in elements (80B: 16B-aligned, conflict-free)

__device__ __forceinline__ unsigned short f2bf(float f) {
    union { float f; unsigned u; } c; c.f = f;
    unsigned r = c.u + 0x7FFFu + ((c.u >> 16) & 1u);
    return (unsigned short)(r >> 16);
}
__device__ __forceinline__ float bf2f(unsigned short u) {
    union { unsigned u; float f; } c; c.u = ((unsigned)u) << 16;
    return c.f;
}
// generic LDS pointer -> 32-bit LDS address (keeps the low 32 bits of the address)
__device__ __forceinline__ unsigned lds_addr32(const void* p) {
    return (unsigned)(unsigned long long)p;
}
// CDNA5 async DMA: global -> LDS, 16B per lane, tracked with ASYNCcnt
__device__ __forceinline__ void async_copy_b128(unsigned lds, const void* gaddr) {
    asm volatile("global_load_async_to_lds_b128 %0, %1, off"
                 :: "v"(lds), "v"(gaddr) : "memory");
}
__device__ __forceinline__ void wait_async0() {
    asm volatile("s_wait_asynccnt 0x0" ::: "memory");
}

// ---------------- f32 -> bf16 conversion ----------------
__global__ void cvt_kernel(const float* __restrict__ in,
                           unsigned short* __restrict__ out, int n) {
    int i = blockIdx.x * blockDim.x + threadIdx.x;
    if (i < n) out[i] = f2bf(in[i]);
}

// ---------------- bf16 WMMA GEMM:  C[M,N] = A[M,K] * Wt[N,K]^T + bias ----------------
// Double-buffered LDS tiles filled by GLOBAL_LOAD_ASYNC_TO_LDS_B128 (ASYNCcnt),
// compute via v_wmma_f32_16x16x32_bf16.
// MODE 0: outb = bf16(acc + bias)
// MODE 1: outf = acc + bias + res          (f32, residual)
// MODE 2: outb = bf16(relu(acc + bias))
template<int MODE>
__global__ __launch_bounds__(256) void gemm_bf16_kernel(
    const unsigned short* __restrict__ A,
    const unsigned short* __restrict__ Wt,
    const float* __restrict__ bias,
    const float* __restrict__ res,
    float* __restrict__ outf,
    unsigned short* __restrict__ outb,
    int M, int N, int K)
{
    __shared__ __align__(16) unsigned short At[2][BM * LDT];
    __shared__ __align__(16) unsigned short Bt[2][BN * LDT];

    const int tid  = threadIdx.x;
    const int wave = tid >> 5;
    const int lane = tid & 31;
    const int wm   = wave & 3;    // 4 waves along M (32 rows each)
    const int wn   = wave >> 2;   // 2 waves along N (64 cols each)
    const int nLoc = lane & 15;   // m/n within 16x16 tile
    const int hi   = lane >> 4;   // lane half

    const int rowBase = blockIdx.y * BM;
    const int colBase = blockIdx.x * BN;

    // cooperative tile loaders: 256 threads, each DMAs 2x16B of A and 2x16B of B
    const int lr = tid >> 1;             // 0..127
    const int lc = (tid & 1) * 16;       // 0 or 16 elements

    const unsigned short* Aptr = A  + (size_t)(rowBase + lr) * K + lc;
    const unsigned short* Bptr = Wt + (size_t)(colBase + lr) * K + lc;

    // per-thread LDS destinations in each buffer
    unsigned ldsA[2], ldsB[2];
    ldsA[0] = lds_addr32(&At[0][lr * LDT + lc]);
    ldsA[1] = lds_addr32(&At[1][lr * LDT + lc]);
    ldsB[0] = lds_addr32(&Bt[0][lr * LDT + lc]);
    ldsB[1] = lds_addr32(&Bt[1][lr * LDT + lc]);

    auto issue_tile = [&](int kk, int q) {
        async_copy_b128(ldsA[q],      Aptr + kk);
        async_copy_b128(ldsA[q] + 16, Aptr + kk + 8);
        async_copy_b128(ldsB[q],      Bptr + kk);
        async_copy_b128(ldsB[q] + 16, Bptr + kk + 8);
    };

    const v8f vzero = {0.f,0.f,0.f,0.f,0.f,0.f,0.f,0.f};
    v8f acc[2][4];
    #pragma unroll
    for (int i = 0; i < 2; ++i)
        #pragma unroll
        for (int j = 0; j < 4; ++j) acc[i][j] = vzero;

    int p = 0;
    issue_tile(0, 0);

    for (int k0 = 0; k0 < K; k0 += BK) {
        wait_async0();         // my DMA for buffer p complete
        __syncthreads();       // everyone's DMA complete; everyone done reading buffer p^1
        if (k0 + BK < K) issue_tile(k0 + BK, p ^ 1);   // overlap DMA with WMMA below

        // A fragment (16x32 bf16): lane half hi holds K = hi*8..+7 and 16+hi*8..+7
        v16bf af[2];
        #pragma unroll
        for (int mt = 0; mt < 2; ++mt) {
            const unsigned short* pa = &At[p][(wm * 32 + mt * 16 + nLoc) * LDT + hi * 8];
            v8u lo = *(const v8u*)pa;
            v8u hh = *(const v8u*)(pa + 16);
            v16u r = __builtin_shufflevector(lo, hh,
                0,1,2,3,4,5,6,7,8,9,10,11,12,13,14,15);
            af[mt] = __builtin_bit_cast(v16bf, r);
        }
        // B fragment (32x16 bf16): lane half hi holds K = hi*8..+7 and 16+hi*8..+7
        v16bf bfm[4];
        #pragma unroll
        for (int nt = 0; nt < 4; ++nt) {
            const unsigned short* pb = &Bt[p][(wn * 64 + nt * 16 + nLoc) * LDT + hi * 8];
            v8u lo = *(const v8u*)pb;
            v8u hh = *(const v8u*)(pb + 16);
            v16u r = __builtin_shufflevector(lo, hh,
                0,1,2,3,4,5,6,7,8,9,10,11,12,13,14,15);
            bfm[nt] = __builtin_bit_cast(v16bf, r);
        }

        #pragma unroll
        for (int mt = 0; mt < 2; ++mt)
            #pragma unroll
            for (int nt = 0; nt < 4; ++nt)
                acc[mt][nt] = __builtin_amdgcn_wmma_f32_16x16x32_bf16(
                    false, af[mt], false, bfm[nt],
                    (short)0, acc[mt][nt], false, false);
        p ^= 1;
    }

    // epilogue: C/D layout -> lane (n = lane&15), vgpr r -> m = hi*8 + r
    #pragma unroll
    for (int mt = 0; mt < 2; ++mt) {
        #pragma unroll
        for (int nt = 0; nt < 4; ++nt) {
            const int col = colBase + wn * 64 + nt * 16 + nLoc;
            const float bv = bias[col];
            #pragma unroll
            for (int r = 0; r < 8; ++r) {
                const int row = rowBase + wm * 32 + mt * 16 + hi * 8 + r;
                const size_t idx = (size_t)row * N + col;
                float v = acc[mt][nt][r] + bv;
                if (MODE == 1)      outf[idx] = v + res[idx];
                else if (MODE == 2) outb[idx] = f2bf(v > 0.f ? v : 0.f);
                else                outb[idx] = f2bf(v);
            }
        }
    }
}

// ---------------- windowed attention: one block per 7-token window ----------------
__global__ __launch_bounds__(128) void attn_kernel(const unsigned short* __restrict__ qkv,
                                                   unsigned short* __restrict__ o)
{
    __shared__ __align__(16) unsigned short sq[7 * 3072];  // whole window's q|k|v (bf16)
    __shared__ float ssc[16 * 49];                         // scores per head

    const int tid = threadIdx.x;
    const size_t base = (size_t)blockIdx.x * 7;

    const unsigned short* src = qkv + base * 3072;
    for (int i = tid; i < (7 * 3072) / 8; i += 128)
        *(v8u*)&sq[i * 8] = *(const v8u*)(src + i * 8);
    __syncthreads();

    // scores: 16 heads x 7x7, scale = 1/sqrt(64)
    for (int e = tid; e < 16 * 49; e += 128) {
        const int h = e / 49, rem = e % 49, qi = rem / 7, ki = rem % 7;
        const unsigned short* qp = &sq[qi * 3072 + h * 64];
        const unsigned short* kp = &sq[ki * 3072 + 1024 + h * 64];
        float s = 0.f;
        #pragma unroll
        for (int d = 0; d < 64; ++d) s += bf2f(qp[d]) * bf2f(kp[d]);
        ssc[e] = s * 0.125f;
    }
    __syncthreads();

    // softmax over each 7-row
    for (int r = tid; r < 16 * 7; r += 128) {
        float* row = &ssc[(r / 7) * 49 + (r % 7) * 7];
        float mx = row[0];
        #pragma unroll
        for (int k = 1; k < 7; ++k) mx = fmaxf(mx, row[k]);
        float sum = 0.f;
        #pragma unroll
        for (int k = 0; k < 7; ++k) { float ev = __expf(row[k] - mx); row[k] = ev; sum += ev; }
        const float inv = 1.f / sum;
        #pragma unroll
        for (int k = 0; k < 7; ++k) row[k] *= inv;
    }
    __syncthreads();

    // o[t, h*64+d] = sum_k attn[h][t][k] * v[k][h*64+d]
    for (int e = tid; e < 7 * 1024; e += 128) {
        const int t = e >> 10, c = e & 1023, h = c >> 6;
        const float* arow = &ssc[h * 49 + t * 7];
        float s = 0.f;
        #pragma unroll
        for (int k = 0; k < 7; ++k) s += arow[k] * bf2f(sq[k * 3072 + 2048 + c]);
        o[(base + t) * 1024 + c] = f2bf(s);
    }
}

// ---------------- LayerNorm: one 256-thread block per row of 1024 ----------------
__global__ __launch_bounds__(256) void ln_kernel(const float* __restrict__ in,
                                                 const float* __restrict__ g,
                                                 const float* __restrict__ b,
                                                 float* __restrict__ outf,
                                                 unsigned short* __restrict__ outb)
{
    __shared__ float rs[256], rs2[256];
    const int tid = threadIdx.x;
    const float* row = in + (size_t)blockIdx.x * HDIM;

    float s = 0.f, s2 = 0.f, vals[4];
    #pragma unroll
    for (int i = 0; i < 4; ++i) {
        float v = row[tid + i * 256];
        vals[i] = v; s += v; s2 += v * v;
    }
    rs[tid] = s; rs2[tid] = s2;
    __syncthreads();
    for (int ofs = 128; ofs > 0; ofs >>= 1) {
        if (tid < ofs) { rs[tid] += rs[tid + ofs]; rs2[tid] += rs2[tid + ofs]; }
        __syncthreads();
    }
    const float mean = rs[0] * (1.f / HDIM);
    const float var  = rs2[0] * (1.f / HDIM) - mean * mean;
    const float rstd = rsqrtf(var + 1e-5f);
    #pragma unroll
    for (int i = 0; i < 4; ++i) {
        const int c = tid + i * 256;
        const float y = (vals[i] - mean) * rstd * g[c] + b[c];
        if (outf) outf[(size_t)blockIdx.x * HDIM + c] = y;
        if (outb) outb[(size_t)blockIdx.x * HDIM + c] = f2bf(y);
    }
}

extern "C" void kernel_launch(void* const* d_in, const int* in_sizes, int n_in,
                              void* d_out, int out_size, void* d_ws, size_t ws_size,
                              hipStream_t stream)
{
    (void)in_sizes; (void)n_in; (void)out_size; (void)ws_size;

    const float* x   = (const float*)d_in[0];
    const float* wq  = (const float*)d_in[1];
    const float* bq  = (const float*)d_in[2];
    const float* wo  = (const float*)d_in[3];
    const float* bo  = (const float*)d_in[4];
    const float* g1  = (const float*)d_in[5];
    const float* b1  = (const float*)d_in[6];
    const float* g2  = (const float*)d_in[7];
    const float* b2  = (const float*)d_in[8];
    const float* w1  = (const float*)d_in[9];
    const float* bb1 = (const float*)d_in[10];
    const float* w2  = (const float*)d_in[11];
    const float* bb2 = (const float*)d_in[12];

    char* ws = (char*)d_ws;
    size_t off = 0;
    auto walloc = [&](size_t bytes) -> char* {
        char* p = ws + off;
        off += (bytes + 255) & ~(size_t)255;
        return p;
    };

    unsigned short* wqkv = (unsigned short*)walloc(3ull * 1024 * 1024 * 2);
    unsigned short* wout = (unsigned short*)walloc(1ull * 1024 * 1024 * 2);
    unsigned short* wff1 = (unsigned short*)walloc(4ull * 1024 * 1024 * 2);
    unsigned short* wff2 = (unsigned short*)walloc(4ull * 1024 * 1024 * 2);
    // regionA: [x_bf16 | qkv_bf16] exactly overlays the later ff1 activation (M*4096 bf16)
    char* regionA = walloc((size_t)MROWS * 4096 * 2);
    unsigned short* xb   = (unsigned short*)regionA;
    unsigned short* qkvb = (unsigned short*)(regionA + (size_t)MROWS * 1024 * 2);
    unsigned short* ff1b = (unsigned short*)regionA;
    unsigned short* obuf = (unsigned short*)walloc((size_t)MROWS * 1024 * 2); // attn out, reused as y_bf16
    float* resf = (float*)walloc((size_t)MROWS * 1024 * 4);
    float* yf   = (float*)walloc((size_t)MROWS * 1024 * 4);

    // bf16 conversions (weights + x)
    cvt_kernel<<<(MROWS * 1024) / 256, 256, 0, stream>>>(x,  xb,   MROWS * 1024);
    cvt_kernel<<<(3 * 1024 * 1024) / 256, 256, 0, stream>>>(wq, wqkv, 3 * 1024 * 1024);
    cvt_kernel<<<(1024 * 1024) / 256, 256, 0, stream>>>(wo, wout, 1024 * 1024);
    cvt_kernel<<<(4 * 1024 * 1024) / 256, 256, 0, stream>>>(w1, wff1, 4 * 1024 * 1024);
    cvt_kernel<<<(4 * 1024 * 1024) / 256, 256, 0, stream>>>(w2, wff2, 4 * 1024 * 1024);

    // 1) QKV projection: [M,1024] x [3072,1024]^T -> bf16 [M,3072]
    gemm_bf16_kernel<0><<<dim3(3072 / BN, MROWS / BM), 256, 0, stream>>>(
        xb, wqkv, bq, nullptr, nullptr, qkvb, MROWS, 3072, 1024);

    // 2) windowed attention -> bf16 [M,1024]
    attn_kernel<<<MROWS / 7, 128, 0, stream>>>(qkvb, obuf);

    // 3) out projection + residual x -> f32 [M,1024]
    gemm_bf16_kernel<1><<<dim3(1024 / BN, MROWS / BM), 256, 0, stream>>>(
        obuf, wout, bo, x, resf, nullptr, MROWS, 1024, 1024);

    // 4) LN1 -> y (f32) + y (bf16, reusing obuf)
    ln_kernel<<<MROWS, 256, 0, stream>>>(resf, g1, b1, yf, obuf);

    // 5) FF1 + ReLU -> bf16 [M,4096]
    gemm_bf16_kernel<2><<<dim3(4096 / BN, MROWS / BM), 256, 0, stream>>>(
        obuf, wff1, bb1, nullptr, nullptr, ff1b, MROWS, 4096, 1024);

    // 6) FF2 + residual y -> f32 [M,1024]
    gemm_bf16_kernel<1><<<dim3(1024 / BN, MROWS / BM), 256, 0, stream>>>(
        ff1b, wff2, bb2, yf, resf, nullptr, MROWS, 1024, 4096);

    // 7) LN2 -> output (f32)
    ln_kernel<<<MROWS, 256, 0, stream>>>(resf, g2, b2, (float*)d_out, nullptr);
}
